// MultiHeadAttention_75110388072814
// MI455X (gfx1250) — matrix operand order repaired and checked by reading, not compiled, not run on hardware
//
#include <hip/hip_runtime.h>
#include <hip/hip_bf16.h>
#include <math.h>

// ---------------------------------------------------------------------------
// MI455X / gfx1250 fused attention block:
//   QKV proj (f16 WMMA, async-LDS double-buffered) -> flash attention
//   (f16 WMMA, f32 online softmax) -> out proj + bias + residual -> LayerNorm.
// ---------------------------------------------------------------------------

typedef __attribute__((ext_vector_type(16))) _Float16 v16h;
typedef __attribute__((ext_vector_type(8)))  _Float16 v8h;
typedef __attribute__((ext_vector_type(8)))  float    v8f;
typedef __attribute__((ext_vector_type(4)))  int      i32x4;

union Frag16 { v16h v; v8h h[2]; };

#define WMMA_F32_F16(a, b, c)                                               \
  __builtin_amdgcn_wmma_f32_16x16x32_f16(false, (a), false, (b), (short)0,  \
                                         (c), false, false)

// ---- async global->LDS path (guarded: falls back to sync staging) ---------
#if defined(__has_builtin)
#if __has_builtin(__builtin_amdgcn_global_load_async_to_lds_b128)
#define ASYNC_LDS 1
#endif
#endif

#define GLOBAL_AS __attribute__((address_space(1)))
#define LDS_AS    __attribute__((address_space(3)))

__device__ inline void stage16(const _Float16* g, _Float16* l) {
#if defined(ASYNC_LDS)
  // GLOBAL_LOAD_ASYNC_TO_LDS_B128: per-lane 16B, tracked on ASYNCcnt.
  // Builtin signature wants a non-const global ptr + LDS ptr.
  // LDS operand: flat address low 32 bits == LDS byte address (aperture rule).
  __builtin_amdgcn_global_load_async_to_lds_b128(
      (GLOBAL_AS i32x4*)(uintptr_t)g,
      (LDS_AS i32x4*)(unsigned)(uintptr_t)l, 0, 0);
#else
  *(v8h*)l = *(const v8h*)g;
#endif
}

__device__ inline void async_wait() {
#if defined(ASYNC_LDS)
#if __has_builtin(__builtin_amdgcn_s_wait_asynccnt)
  __builtin_amdgcn_s_wait_asynccnt(0);
#else
  asm volatile("s_wait_asynccnt 0" ::: "memory");
#endif
#endif
}

static constexpr int Bc = 2, Sc = 2048, Hc = 1024, NHc = 16, HDc = 64;
static constexpr int Mrows = Bc * Sc;   // 4096
static constexpr int Kdim  = Hc;        // 1024
static constexpr int Ndim  = Hc;        // 1024

__device__ inline v8f vzero8() {
  v8f z;
#pragma unroll
  for (int i = 0; i < 8; ++i) z[i] = 0.0f;
  return z;
}

// ---------------------------------------------------------------------------
// fp32 -> f16 conversion (grid-stride)
// ---------------------------------------------------------------------------
__global__ void k_cvt(const float* __restrict__ src,
                      _Float16* __restrict__ dst, int n) {
  int i = blockIdx.x * blockDim.x + threadIdx.x;
  int stride = gridDim.x * blockDim.x;
  for (; i < n; i += stride) dst[i] = (_Float16)src[i];
}

// ---------------------------------------------------------------------------
// GEMM: C[M,N] = A[M,K] * W[N,K]^T   (torch Linear: y = x @ W.T)
// Block tile 128x128, 8 waves, each wave owns a 32x64 sub-tile (2x4 WMMAs).
// LDS double-buffered; tiles staged with GLOBAL_LOAD_ASYNC_TO_LDS_B128.
// mode 0: store f16, head layout [B,NH,S,HD]            (Q, K)
// mode 2: store f16, transposed head layout [B,NH,HD,S] (V)
// mode 1: store f32 = acc + bias[n] + resid[m,n]        (out proj)
// ---------------------------------------------------------------------------
__global__ __launch_bounds__(256) void k_gemm(
    const _Float16* __restrict__ A, const _Float16* __restrict__ W, int mode,
    _Float16* __restrict__ outH, float* __restrict__ outF,
    const float* __restrict__ bias, const float* __restrict__ resid) {
  __shared__ alignas(16) _Float16 As[2][128][40];  // +8 pad: conflict-free
  __shared__ alignas(16) _Float16 Ws[2][128][40];

  const int tid  = threadIdx.x;
  const int lane = tid & 31;
  const int w    = tid >> 5;
  const int nBase = blockIdx.x * 128;
  const int mBase = blockIdx.y * 128;
  const int mOff = (w >> 1) * 32;   // wave row: 4 rows of 32
  const int nOff = (w & 1) * 64;    // wave col: 2 cols of 64
  const int nl = lane & 15;
  const int hiHalf = lane >> 4;
  const int aK0 = hiHalf ? 8 : 0;   // A-frag k base (ISA 16-bit A layout)
  const int bK0 = hiHalf ? 8 : 0;   // B-frag k base (same order as A)

  v8f acc[2][4];
#pragma unroll
  for (int i = 0; i < 2; ++i)
#pragma unroll
    for (int j = 0; j < 4; ++j) acc[i][j] = vzero8();

  const int ldRow = tid >> 1;          // 128 rows, 2 threads per row
  const int ldSeg = (tid & 1) * 16;    // each thread: 16 halves (2 x b128)

  // stage one 128x32 A-tile + 128x32 W-tile into buffer `buf`
  auto stage = [&](int buf, int k0) {
    const _Float16* ag = A + (size_t)(mBase + ldRow) * Kdim + k0 + ldSeg;
    stage16(ag,     &As[buf][ldRow][ldSeg]);
    stage16(ag + 8, &As[buf][ldRow][ldSeg + 8]);
    const _Float16* wg = W + (size_t)(nBase + ldRow) * Kdim + k0 + ldSeg;
    stage16(wg,     &Ws[buf][ldRow][ldSeg]);
    stage16(wg + 8, &Ws[buf][ldRow][ldSeg + 8]);
  };

  stage(0, 0);
  async_wait();
  __syncthreads();

  int cur = 0;
  for (int k0 = 0; k0 < Kdim; k0 += 32) {
    // prefetch next tile into the other buffer while we compute
    if (k0 + 32 < Kdim) stage(cur ^ 1, k0 + 32);

    Frag16 a[2], b[4];
#pragma unroll
    for (int i = 0; i < 2; ++i) {
      const _Float16* p = &As[cur][mOff + i * 16 + nl][0];
      a[i].h[0] = *(const v8h*)(p + aK0);       // k = aK0..aK0+7
      a[i].h[1] = *(const v8h*)(p + aK0 + 16);  // k = aK0+16..aK0+23
    }
#pragma unroll
    for (int j = 0; j < 4; ++j) {
      const _Float16* p = &Ws[cur][nOff + j * 16 + nl][0];
      b[j].h[0] = *(const v8h*)(p + bK0);       // k = bK0..bK0+7
      b[j].h[1] = *(const v8h*)(p + bK0 + 16);  // k = bK0+16..bK0+23
    }
#pragma unroll
    for (int i = 0; i < 2; ++i)
#pragma unroll
      for (int j = 0; j < 4; ++j)
        acc[i][j] = WMMA_F32_F16(a[i].v, b[j].v, acc[i][j]);

    async_wait();     // prefetch landed (own wave's ASYNCcnt)
    __syncthreads();  // all waves done reading `cur`, all data visible
    cur ^= 1;
  }

  // Epilogue: C layout -> lane nl = col, row = vgpr + 8*hiHalf
#pragma unroll
  for (int i = 0; i < 2; ++i)
#pragma unroll
    for (int j = 0; j < 4; ++j)
#pragma unroll
      for (int r = 0; r < 8; ++r) {
        const int m = mBase + mOff + i * 16 + hiHalf * 8 + r;
        const int n = nBase + nOff + j * 16 + nl;
        const float val = acc[i][j][r];
        if (mode == 0) {  // [B,NH,S,HD]
          const int bb = m >> 11, s = m & (Sc - 1);
          const int hd = n >> 6, d = n & (HDc - 1);
          outH[(((size_t)bb * NHc + hd) * Sc + s) * HDc + d] = (_Float16)val;
        } else if (mode == 2) {  // [B,NH,HD,S]
          const int bb = m >> 11, s = m & (Sc - 1);
          const int hd = n >> 6, d = n & (HDc - 1);
          outH[(((size_t)bb * NHc + hd) * HDc + d) * Sc + s] = (_Float16)val;
        } else {  // mode 1: f32 + bias + residual
          outF[(size_t)m * Ndim + n] =
              val + bias[n] + resid[(size_t)m * Ndim + n];
        }
      }
}

// ---------------------------------------------------------------------------
// Flash attention. One wave per (b, head, 16-query-row tile).
// Q kept in registers (2 A-frags over d). Loop over 32-key blocks:
//   4 WMMAs -> 16x32 scores, scale+mask, online softmax (half-wave shuffles),
//   P through private LDS slice (C-layout -> A-layout), 4 PV WMMAs vs V^T.
// ---------------------------------------------------------------------------
__global__ __launch_bounds__(256) void k_attn(
    const _Float16* __restrict__ Q, const _Float16* __restrict__ Kh,
    const _Float16* __restrict__ Vt, const int* __restrict__ mask,
    _Float16* __restrict__ ctx) {
  __shared__ alignas(16) _Float16 Pb[8][16][40];  // per-wave P staging

  const int tid  = threadIdx.x;
  const int lane = tid & 31;
  const int w    = tid >> 5;
  const int gw   = blockIdx.x * 8 + w;   // 4096 waves total
  const int bh   = gw >> 7;              // (b*NH + head), 0..31
  const int qt   = gw & 127;             // query tile, 0..127
  const int b    = bh >> 4;
  const int head = bh & 15;
  const int nl = lane & 15;
  const int hiHalf = lane >> 4;
  const int aK0 = hiHalf ? 8 : 0;
  const int bK0 = hiHalf ? 8 : 0;

  // Q A-fragments: row = qt*16 + nl, contraction d in two 32-wide halves
  Frag16 qa[2];
  {
    const _Float16* qp = Q + ((size_t)bh * Sc + qt * 16 + nl) * HDc;
#pragma unroll
    for (int i = 0; i < 2; ++i) {
      qa[i].h[0] = *(const v8h*)(qp + i * 32 + aK0);
      qa[i].h[1] = *(const v8h*)(qp + i * 32 + aK0 + 16);
    }
  }

  float mrun[8], lrun[8];
#pragma unroll
  for (int r = 0; r < 8; ++r) { mrun[r] = -3.0e38f; lrun[r] = 0.0f; }
  v8f o[4];
#pragma unroll
  for (int j = 0; j < 4; ++j) o[j] = vzero8();

  const size_t maskBase = (size_t)b * Sc * Sc;  // mask[b, :, :]

  for (int sB = 0; sB < Sc; sB += 32) {
    // prefetch next block's mask stream (most latency-exposed accesses)
    if (sB + 32 < Sc)
      __builtin_prefetch(
          &mask[maskBase + (size_t)(qt * 16 + hiHalf * 8) * Sc + sB + 32 + nl],
          0, 0);

    // ---- scores: two 16x16 tiles over keys [sB, sB+32)
    v8f sc[2];
#pragma unroll
    for (int t = 0; t < 2; ++t) {
      Frag16 kf0, kf1;  // B-frags: col = key, k = d
      const _Float16* kp = Kh + ((size_t)bh * Sc + sB + t * 16 + nl) * HDc;
      kf0.h[0] = *(const v8h*)(kp + bK0);            // d = bK0..+7
      kf0.h[1] = *(const v8h*)(kp + bK0 + 16);
      kf1.h[0] = *(const v8h*)(kp + 32 + bK0);
      kf1.h[1] = *(const v8h*)(kp + 32 + bK0 + 16);
      sc[t] = WMMA_F32_F16(qa[0].v, kf0.v, vzero8());
      sc[t] = WMMA_F32_F16(qa[1].v, kf1.v, sc[t]);
    }

    // ---- scale 1/sqrt(64) + mask (matches reference -1e9 semantics)
#pragma unroll
    for (int t = 0; t < 2; ++t) {
      const int key = sB + t * 16 + nl;
#pragma unroll
      for (int r = 0; r < 8; ++r) {
        const int qrow = qt * 16 + hiHalf * 8 + r;
        float x = sc[t][r] * 0.125f;
        if (mask[maskBase + (size_t)qrow * Sc + key] == 0) x = -1.0e9f;
        sc[t][r] = x;
      }
    }

    // ---- online softmax (rows live in 16-lane halves -> xor off < 16)
    float p0v[8], p1v[8];
#pragma unroll
    for (int r = 0; r < 8; ++r) {
      float rm = fmaxf(sc[0][r], sc[1][r]);
#pragma unroll
      for (int off = 1; off < 16; off <<= 1)
        rm = fmaxf(rm, __shfl_xor(rm, off, 32));
      const float mNew = fmaxf(mrun[r], rm);
      const float corr = __expf(mrun[r] - mNew);
      const float p0 = __expf(sc[0][r] - mNew);
      const float p1 = __expf(sc[1][r] - mNew);
      float rs = p0 + p1;
#pragma unroll
      for (int off = 1; off < 16; off <<= 1) rs += __shfl_xor(rs, off, 32);
      lrun[r] = lrun[r] * corr + rs;
      mrun[r] = mNew;
#pragma unroll
      for (int j = 0; j < 4; ++j) o[j][r] *= corr;
      p0v[r] = p0; p1v[r] = p1;
    }

    // ---- P: C-layout -> A-layout via private LDS slice (same-wave DS ops
    //      are processed in order; no cross-wave sharing -> no barrier)
#pragma unroll
    for (int r = 0; r < 8; ++r) {
      Pb[w][hiHalf * 8 + r][nl]      = (_Float16)p0v[r];
      Pb[w][hiHalf * 8 + r][16 + nl] = (_Float16)p1v[r];
    }
    Frag16 pa;
    {
      const _Float16* pp = &Pb[w][nl][0];
      pa.h[0] = *(const v8h*)(pp + aK0);
      pa.h[1] = *(const v8h*)(pp + aK0 + 16);
    }

    // ---- PV: O[16q x 64d] += P[16x32] * V[32k x 64d] (V stored d-major)
#pragma unroll
    for (int j = 0; j < 4; ++j) {
      Frag16 vf;  // B-frag: col = d, k = key (contiguous in V^T)
      const _Float16* vp =
          Vt + ((size_t)bh * HDc + j * 16 + nl) * Sc + sB + bK0;
      vf.h[0] = *(const v8h*)vp;
      vf.h[1] = *(const v8h*)(vp + 16);
      o[j] = WMMA_F32_F16(pa.v, vf.v, o[j]);
    }
  }

  // ---- normalize + store ctx [B,S,H] (heads re-merged)
#pragma unroll
  for (int r = 0; r < 8; ++r) {
    const float inv = 1.0f / lrun[r];  // lrun >= 1 always (exp(max-max)=1)
    const int s = qt * 16 + hiHalf * 8 + r;
#pragma unroll
    for (int j = 0; j < 4; ++j)
      ctx[((size_t)b * Sc + s) * Hc + head * HDc + j * 16 + nl] =
          (_Float16)(o[j][r] * inv);
  }
}

// ---------------------------------------------------------------------------
// LayerNorm over H=1024, one block per row, eps = 1e-6
// ---------------------------------------------------------------------------
__global__ __launch_bounds__(256) void k_ln(const float* __restrict__ x,
                                            const float* __restrict__ gamma,
                                            const float* __restrict__ beta,
                                            float* __restrict__ out) {
  __shared__ float rs[256], rq[256];
  const int row = blockIdx.x;
  const float* xr = x + (size_t)row * Hc;
  float s = 0.0f, s2 = 0.0f;
  for (int i = threadIdx.x; i < Hc; i += 256) {
    const float v = xr[i];
    s += v; s2 += v * v;
  }
  rs[threadIdx.x] = s; rq[threadIdx.x] = s2;
  __syncthreads();
  for (int off = 128; off > 0; off >>= 1) {
    if (threadIdx.x < off) {
      rs[threadIdx.x] += rs[threadIdx.x + off];
      rq[threadIdx.x] += rq[threadIdx.x + off];
    }
    __syncthreads();
  }
  const float mu   = rs[0] * (1.0f / Hc);
  const float var  = rq[0] * (1.0f / Hc) - mu * mu;
  const float rstd = rsqrtf(var + 1e-6f);
  for (int i = threadIdx.x; i < Hc; i += 256)
    out[(size_t)row * Hc + i] = (xr[i] - mu) * rstd * gamma[i] + beta[i];
}

// ---------------------------------------------------------------------------
// Host-side launch
// ---------------------------------------------------------------------------
extern "C" void kernel_launch(void* const* d_in, const int* in_sizes, int n_in,
                              void* d_out, int out_size, void* d_ws,
                              size_t ws_size, hipStream_t stream) {
  const float* hs    = (const float*)d_in[0];  // [B,S,H]
  const int*   mask  = (const int*)d_in[1];    // [B,S,S]
  const float* Wq    = (const float*)d_in[2];
  const float* Wk    = (const float*)d_in[3];
  const float* Wv    = (const float*)d_in[4];
  const float* Wd    = (const float*)d_in[5];
  const float* bd    = (const float*)d_in[6];
  const float* gamma = (const float*)d_in[7];
  const float* beta  = (const float*)d_in[8];
  float* outp = (float*)d_out;

  // Workspace layout (halves unless noted); total 64 MB.
  _Float16* wsH = (_Float16*)d_ws;
  _Float16* hf  = wsH;                         //  4,194,304 h : hidden f16
  _Float16* wqh = wsH + 4194304;               //  1,048,576 h
  _Float16* wkh = wsH + 5242880;
  _Float16* wvh = wsH + 6291456;
  _Float16* wdh = wsH + 7340032;
  _Float16* Qh  = wsH + 8388608;               //  [B,NH,S,HD]
  _Float16* Khd = wsH + 12582912;              //  [B,NH,S,HD]
  _Float16* Vt  = wsH + 16777216;              //  [B,NH,HD,S]
  _Float16* ctx = wsH + 20971520;              //  [B*S, H]
  float* proj   = (float*)((char*)d_ws + 50331648);  // [B*S, H] f32

  // 1) convert to f16
  k_cvt<<<2048, 256, 0, stream>>>(hs, hf, Mrows * Hc);
  k_cvt<<<512, 256, 0, stream>>>(Wq, wqh, Hc * Hc);
  k_cvt<<<512, 256, 0, stream>>>(Wk, wkh, Hc * Hc);
  k_cvt<<<512, 256, 0, stream>>>(Wv, wvh, Hc * Hc);
  k_cvt<<<512, 256, 0, stream>>>(Wd, wdh, Hc * Hc);

  // 2) Q/K/V projections
  dim3 gg(Ndim / 128, Mrows / 128);  // (8, 32)
  k_gemm<<<gg, 256, 0, stream>>>(hf, wqh, 0, Qh, nullptr, nullptr, nullptr);
  k_gemm<<<gg, 256, 0, stream>>>(hf, wkh, 0, Khd, nullptr, nullptr, nullptr);
  k_gemm<<<gg, 256, 0, stream>>>(hf, wvh, 2, Vt, nullptr, nullptr, nullptr);

  // 3) flash attention: 4096 waves = 512 blocks x 8 waves
  k_attn<<<512, 256, 0, stream>>>(Qh, Khd, Vt, mask, ctx);

  // 4) out projection + bias + residual (f32)
  k_gemm<<<gg, 256, 0, stream>>>(ctx, wdh, 1, nullptr, proj, bd, hs);

  // 5) layernorm
  k_ln<<<Mrows, 256, 0, stream>>>(proj, gamma, beta, outp);
}
